// SurfaceModel_35278861369953
// MI455X (gfx1250) — matrix operand order repaired and checked by reading, not compiled, not run on hardware
//
#include <hip/hip_runtime.h>
#include <hip/hip_bf16.h>

typedef __attribute__((ext_vector_type(16))) __bf16 v16bf;
typedef __attribute__((ext_vector_type(8)))  __bf16 v8bf;
typedef __attribute__((ext_vector_type(8)))  float  v8f;

#define TWO_PI_F  6.28318530717958647692f
#define INV_4PI_F 0.07957747154594766788f

static_assert(sizeof(__bf16) == 2, "bf16 must be 2 bytes");

// ---------------- fast math helpers (native VALU trans ops) ----------------
__device__ __forceinline__ float fast_exp(float x) {  // e^x via v_exp_f32 (2^x)
  return __builtin_amdgcn_exp2f(x * 1.44269504088896341f);
}
__device__ __forceinline__ float fast_log(float x) {  // ln(x) via v_log_f32 (log2)
  return __builtin_amdgcn_logf(x) * 0.69314718055994531f;
}
// hardware v_sin_f32 / v_cos_f32 take *turns*: sin(2*pi*t). Our proj = 2*pi*t,
// so feeding t directly gives sin(proj)/cos(proj) with no extra multiply.
__device__ __forceinline__ float sin2pi(float t) { return __builtin_amdgcn_sinf(t); }
__device__ __forceinline__ float cos2pi(float t) { return __builtin_amdgcn_cosf(t); }

// ---------------- WMMA helpers (CDNA5 bf16 16x16x32, f32 accum) ------------
__device__ __forceinline__ v8f wmma_bf16(v16bf a, v16bf b, v8f c) {
  return __builtin_amdgcn_wmma_f32_16x16x32_bf16(false, a, false, b, (short)0, c,
                                                 false, false);
}

// A fragment from LDS (row-major, given stride in elements).
// k order used here (same as loadB_g): lane L -> M = L%16; element e -> K =
// (L/16)*16 + e  => one contiguous 16-element run, loaded as two 16B halves.
__device__ __forceinline__ v16bf loadA_lds(const __bf16* h, int stride, int row,
                                           int kc, int half) {
  const __bf16* base = h + row * stride + kc * 32 + half * 16;
  v8bf lo = *(const v8bf*)(base);
  v8bf hi = *(const v8bf*)(base + 8);
  return __builtin_shufflevector(lo, hi, 0, 1, 2, 3, 4, 5, 6, 7,
                                 8, 9, 10, 11, 12, 13, 14, 15);
}

// B fragment from global weights, K contiguous in memory for fixed n.
// k order used here (same as loadA_lds): lane L -> N = L%16; element e -> K = (L/16)*16 + e.
__device__ __forceinline__ v16bf loadB_g(const __bf16* W, int kdim, int n, int kc, int half) {
  return *(const v16bf*)(W + n * kdim + kc * 32 + half * 16);
}

__device__ __forceinline__ void zacc(v8f acc[2][4]) {
  v8f z = {0.f, 0.f, 0.f, 0.f, 0.f, 0.f, 0.f, 0.f};
#pragma unroll
  for (int i = 0; i < 2; ++i)
#pragma unroll
    for (int j = 0; j < 4; ++j) acc[i][j] = z;
}

// 32xK x Kx256 GEMM step: 2 row-tiles x 4 col-tiles per wave.
__device__ __forceinline__ void gemm256(const __bf16* A, const __bf16* W,
                                        int colW, int l16, int half, v8f acc[2][4]) {
#pragma unroll
  for (int kc = 0; kc < 8; ++kc) {
    v16bf a0 = loadA_lds(A, 256, l16, kc, half);
    v16bf a1 = loadA_lds(A, 256, 16 + l16, kc, half);
#pragma unroll
    for (int ct = 0; ct < 4; ++ct) {
      v16bf b = loadB_g(W, 256, colW + ct * 16 + l16, kc, half);
      acc[0][ct] = wmma_bf16(a0, b, acc[0][ct]);
      acc[1][ct] = wmma_bf16(a1, b, acc[1][ct]);
    }
  }
}

// forward epilogue: z = acc + bias; h = softplus(z); sg = sigmoid(z)
// C layout: element r of lane L -> row = r + 8*(L/16) (+16*rowTile), col = L%16.
__device__ __forceinline__ void fwd_epi(v8f acc[2][4], const float* bias,
                                        __bf16* hOut, __bf16* sgOut,
                                        int colW, int l16, int half) {
#pragma unroll
  for (int rt = 0; rt < 2; ++rt)
#pragma unroll
    for (int ct = 0; ct < 4; ++ct) {
      int colG = colW + ct * 16 + l16;
      float bb = bias[colG];
#pragma unroll
      for (int r = 0; r < 8; ++r) {
        int row = rt * 16 + 8 * half + r;
        float z = acc[rt][ct][r] + bb;
        float ez = fast_exp(z);
        float e1 = 1.f + ez;
        bool big = z > 15.f;
        float sp = big ? z : fast_log(e1);
        float sgv = big ? 1.f : ez * __builtin_amdgcn_rcpf(e1);
        hOut[row * 256 + colG] = (__bf16)sp;
        sgOut[row * 256 + colG] = (__bf16)sgv;
      }
    }
}

// backward epilogue: dz_prev = acc * sg_prev
__device__ __forceinline__ void bwd_epi(v8f acc[2][4], const __bf16* sgPrev,
                                        __bf16* gOut, int colW, int l16, int half) {
#pragma unroll
  for (int rt = 0; rt < 2; ++rt)
#pragma unroll
    for (int ct = 0; ct < 4; ++ct) {
      int colG = colW + ct * 16 + l16;
#pragma unroll
      for (int r = 0; r < 8; ++r) {
        int row = rt * 16 + 8 * half + r;
        float v = acc[rt][ct][r] * (float)sgPrev[row * 256 + colG];
        gOut[row * 256 + colG] = (__bf16)v;
      }
    }
}

// ---------------- prep kernel: fp32 -> bf16 (plain + transposed) -----------
__global__ void cvt_kernel(const float* __restrict__ src, __bf16* __restrict__ dstB,
                           __bf16* __restrict__ dstT, int fi, int fo) {
  int idx = blockIdx.x * blockDim.x + threadIdx.x;
  if (idx >= fi * fo) return;
  int k = idx / fo;
  int n = idx - k * fo;
  __bf16 v = (__bf16)src[idx];
  dstB[idx] = v;            // [fi, fo]  (backward: B[k,n] = W^T[k,n] = W[n,k])
  dstT[n * fi + k] = v;     // [fo, fi]  (forward:  B[k,n] = W[k,n], K contiguous)
}

// ---------------- Biot-Savart kernel (VALU, trivial FLOPs) -----------------
__global__ __launch_bounds__(256) void biot_kernel(const float* __restrict__ x,
                                                   const float* __restrict__ bdry,
                                                   float* __restrict__ alphaOut,
                                                   int N, int M) {
  __shared__ float sB[1536];  // M<=512 loop points, 3 floats each
  for (int i = threadIdx.x; i < M * 3; i += 256) sB[i] = bdry[i];
  __syncthreads();
  int i = blockIdx.x * 256 + threadIdx.x;
  if (i >= N) return;
  float px = fminf(1.f, fmaxf(-1.f, x[i * 3 + 0]));
  float py = fminf(1.f, fmaxf(-1.f, x[i * 3 + 1]));
  float pz = fminf(1.f, fmaxf(-1.f, x[i * 3 + 2]));
  float ax = 0.f, ay = 0.f, az = 0.f;
  for (int m = 0; m < M; ++m) {
    int m2 = (m + 1 == M) ? 0 : m + 1;
    float b0x = sB[m * 3 + 0], b0y = sB[m * 3 + 1], b0z = sB[m * 3 + 2];
    float b1x = sB[m2 * 3 + 0], b1y = sB[m2 * 3 + 1], b1z = sB[m2 * 3 + 2];
    float sx = b1x - b0x, sy = b1y - b0y, sz = b1z - b0z;
    float mx = 0.5f * (b1x + b0x), my = 0.5f * (b1y + b0y), mz = 0.5f * (b1z + b0z);
    float dx = px - mx, dy = py - my, dz = pz - mz;
    float cx = sy * dz - sz * dy;
    float cy = sz * dx - sx * dz;
    float cz = sx * dy - sy * dx;
    float r2 = dx * dx + dy * dy + dz * dz;
    float ir = __builtin_amdgcn_rsqf(r2);
    float ir3 = ir * ir * ir;
    ax += cx * ir3; ay += cy * ir3; az += cz * ir3;
  }
  alphaOut[i * 3 + 0] = ax * INV_4PI_F;
  alphaOut[i * 3 + 1] = ay * INV_4PI_F;
  alphaOut[i * 3 + 2] = az * INV_4PI_F;
}

// ---------------- LDS layout for the main kernel ---------------------------
#define OFF_SBRFF 0        // 3072 f32  = 12288 B
#define OFF_SBIAS 12288    // 1024 f32  = 4096 B
#define OFF_SW5   16384    // 256 f32   = 1024 B
#define OFF_SX    17408    // 96 f32    = 384 B
#define OFF_SDF   17792    // 96 f32    = 384 B
#define OFF_SF    18176    // 32 f32    = 128 B
#define OFF_SY    18304    // 32*2048 bf16 = 131072 B  (32B aligned)
#define OFF_HA    149376   // 8192 bf16 = 16384 B
#define OFF_HB    165760   // 8192 bf16 = 16384 B
#define OFF_SG    182144   // 4*8192 bf16 = 65536 B
#define SMEM_BYTES 247680  // ~242 KB of the 320 KB WGP pool

// ---------------- main fused MLP forward + input-gradient kernel -----------
__global__ __launch_bounds__(128) void mlp_kernel(
    const float* __restrict__ x, const float* __restrict__ Brff,
    const float* __restrict__ b1, const float* __restrict__ b2,
    const float* __restrict__ b3, const float* __restrict__ b4,
    const float* __restrict__ W5, const float* __restrict__ b5p,
    const __bf16* __restrict__ W1t, const __bf16* __restrict__ W1b,
    const __bf16* __restrict__ W2t, const __bf16* __restrict__ W2b,
    const __bf16* __restrict__ W3t, const __bf16* __restrict__ W3b,
    const __bf16* __restrict__ W4t, const __bf16* __restrict__ W4b,
    float* __restrict__ out, int N) {
  extern __shared__ char smem[];
  float*  sBrff = (float*)(smem + OFF_SBRFF);
  float*  sBias = (float*)(smem + OFF_SBIAS);
  float*  sW5   = (float*)(smem + OFF_SW5);
  float*  sX    = (float*)(smem + OFF_SX);
  float*  sDf   = (float*)(smem + OFF_SDF);
  float*  sF    = (float*)(smem + OFF_SF);
  __bf16* sY    = (__bf16*)(smem + OFF_SY);   // [32][2048]: sin | cos
  __bf16* hA    = (__bf16*)(smem + OFF_HA);
  __bf16* hB    = (__bf16*)(smem + OFF_HB);
  __bf16* sg    = (__bf16*)(smem + OFF_SG);

  const int tid  = threadIdx.x;
  const int wave = tid >> 5;
  const int lane = tid & 31;
  const int half = lane >> 4;
  const int l16  = lane & 15;
  const int colW = wave * 64;          // forward/backward-mid col strip per wave
  const int r0   = blockIdx.x * 32;    // 32 rows per workgroup

  // ---- stage constants + clipped x into LDS ----
  for (int i = tid; i < 3072; i += 128) sBrff[i] = Brff[i];
  for (int i = tid; i < 256; i += 128) {
    sBias[i]       = b1[i];
    sBias[256 + i] = b2[i];
    sBias[512 + i] = b3[i];
    sBias[768 + i] = b4[i];
    sW5[i]         = W5[i];
  }
  if (tid < 96) {
    float v = x[r0 * 3 + tid];
    sX[tid] = fminf(1.f, fmaxf(-1.f, v));
    sDf[tid] = 0.f;
  }
  if (tid < 32) sF[tid] = b5p[0];
  __syncthreads();

  // ---- precompute RFF features y = [sin(2pi t) | cos(2pi t)] once per block ----
  // shared by all 4 waves for the layer-1 A fragments AND reused by the
  // backward chain rule (d/dt sin = stored cos column; d/dt cos = -stored sin).
  for (int i = tid; i < 32 * 1024; i += 128) {
    int row = i >> 10;
    int col = i & 1023;
    float t = sX[row * 3 + 0] * sBrff[col] +
              sX[row * 3 + 1] * sBrff[1024 + col] +
              sX[row * 3 + 2] * sBrff[2048 + col];
    sY[row * 2048 + col]        = (__bf16)sin2pi(t);
    sY[row * 2048 + 1024 + col] = (__bf16)cos2pi(t);
  }
  __syncthreads();

  v8f acc[2][4];

  // ---- layer 1: y[32,2048] @ W1 -> h1 [32,256] ----
  {
    zacc(acc);
    for (int kc = 0; kc < 64; ++kc) {
      v16bf a0 = loadA_lds(sY, 2048, l16, kc, half);
      v16bf a1 = loadA_lds(sY, 2048, 16 + l16, kc, half);
#pragma unroll
      for (int ct = 0; ct < 4; ++ct) {
        v16bf b = loadB_g(W1t, 2048, colW + ct * 16 + l16, kc, half);
        acc[0][ct] = wmma_bf16(a0, b, acc[0][ct]);
        acc[1][ct] = wmma_bf16(a1, b, acc[1][ct]);
      }
    }
    fwd_epi(acc, sBias + 0, hA, sg + 0 * 8192, colW, l16, half);
  }
  __syncthreads();

  // ---- layers 2..4 ----
  zacc(acc); gemm256(hA, W2t, colW, l16, half, acc);
  fwd_epi(acc, sBias + 256, hB, sg + 1 * 8192, colW, l16, half);
  __syncthreads();
  zacc(acc); gemm256(hB, W3t, colW, l16, half, acc);
  fwd_epi(acc, sBias + 512, hA, sg + 2 * 8192, colW, l16, half);
  __syncthreads();
  zacc(acc); gemm256(hA, W4t, colW, l16, half, acc);
  fwd_epi(acc, sBias + 768, hB, sg + 3 * 8192, colW, l16, half);
  __syncthreads();

  // ---- layer 5 (f = h4 @ W5 + b5, parallel reduce) and dz4 = W5 * sg4 ----
  {
    int row = tid & 31;
    int seg = tid >> 5;  // 4 segments of 64 each
    float s = 0.f;
    for (int k = seg * 64; k < seg * 64 + 64; ++k)
      s += (float)hB[row * 256 + k] * sW5[k];
    atomicAdd(&sF[row], s);
  }
  for (int i = tid; i < 8192; i += 128) {
    int c = i & 255;
    hA[i] = (__bf16)(sW5[c] * (float)sg[3 * 8192 + i]);  // dz4 -> hA
  }
  __syncthreads();
  if (tid < 32) out[r0 + tid] = sF[tid];  // f region: [0, N)

  // ---- backward chain: dz_{l-1} = (dz_l @ W_l^T) * sg_{l-1} ----
  zacc(acc); gemm256(hA, W4b, colW, l16, half, acc);
  bwd_epi(acc, sg + 2 * 8192, hB, colW, l16, half);   // dz3 -> hB
  __syncthreads();
  zacc(acc); gemm256(hB, W3b, colW, l16, half, acc);
  bwd_epi(acc, sg + 1 * 8192, hA, colW, l16, half);   // dz2 -> hA
  __syncthreads();
  zacc(acc); gemm256(hA, W2b, colW, l16, half, acc);
  bwd_epi(acc, sg + 0 * 8192, hB, colW, l16, half);   // dz1 -> hB
  __syncthreads();

  // ---- fused: dy = dz1 @ W1^T [32,2048]; df = 2pi * B @ (dy_s*cos - dy_c*sin) ----
  // derivative factors come straight from the stored y matrix (no trig here):
  //   col < 1024 (sin feature):  d/dt = +cos  = sY[row][1024+col]
  //   col >= 1024 (cos feature): d/dt = -sin  = -sY[row][col-1024]
  for (int rt = 0; rt < 2; ++rt) {
    float dfp[24];
#pragma unroll
    for (int i = 0; i < 24; ++i) dfp[i] = 0.f;
    for (int ct = 0; ct < 32; ++ct) {
      int colG = wave * 512 + ct * 16 + l16;  // column of dy in [0,2048)
      v8f a = {0.f, 0.f, 0.f, 0.f, 0.f, 0.f, 0.f, 0.f};
#pragma unroll
      for (int kc = 0; kc < 8; ++kc) {
        v16bf af = loadA_lds(hB, 256, rt * 16 + l16, kc, half);
        v16bf bf = loadB_g(W1b, 256, colG, kc, half);
        a = wmma_bf16(af, bf, a);
      }
      bool isSin = (colG < 1024);
      int colD   = isSin ? (1024 + colG) : (colG - 1024);
      float sgn  = isSin ? 1.f : -1.f;
      int col    = colG & 1023;
      float B0 = sBrff[col], B1 = sBrff[1024 + col], B2 = sBrff[2048 + col];
#pragma unroll
      for (int r = 0; r < 8; ++r) {
        int row = rt * 16 + 8 * half + r;
        float d = sgn * a[r] * (float)sY[row * 2048 + colD];
        dfp[r * 3 + 0] += d * B0;
        dfp[r * 3 + 1] += d * B1;
        dfp[r * 3 + 2] += d * B2;
      }
    }
#pragma unroll
    for (int r = 0; r < 8; ++r) {
      int row = rt * 16 + 8 * half + r;
      atomicAdd(&sDf[row * 3 + 0], dfp[r * 3 + 0]);
      atomicAdd(&sDf[row * 3 + 1], dfp[r * 3 + 1]);
      atomicAdd(&sDf[row * 3 + 2], dfp[r * 3 + 2]);
    }
  }
  __syncthreads();

  // ---- write df and current = df + alpha ----
  if (tid < 96) {
    float v = sDf[tid] * TWO_PI_F;
    long gi = (long)r0 * 3 + tid;
    out[(long)4 * N + gi] = v;                          // df region
    out[(long)N + gi] = v + out[(long)7 * N + gi];      // current = df + alpha
  }
}

// ---------------- host-side launcher ---------------------------------------
extern "C" void kernel_launch(void* const* d_in, const int* in_sizes, int n_in,
                              void* d_out, int out_size, void* d_ws, size_t ws_size,
                              hipStream_t stream) {
  const float* x    = (const float*)d_in[0];
  const float* bdry = (const float*)d_in[1];
  const float* Brff = (const float*)d_in[2];
  const float* W1   = (const float*)d_in[3];
  const float* b1   = (const float*)d_in[4];
  const float* W2   = (const float*)d_in[5];
  const float* b2   = (const float*)d_in[6];
  const float* W3   = (const float*)d_in[7];
  const float* b3   = (const float*)d_in[8];
  const float* W4   = (const float*)d_in[9];
  const float* b4   = (const float*)d_in[10];
  const float* W5   = (const float*)d_in[11];
  const float* b5   = (const float*)d_in[12];
  float* out = (float*)d_out;

  const int N = in_sizes[0] / 3;   // 65536 sample points
  const int M = in_sizes[1] / 3;   // 512 boundary points

  // workspace: bf16 weight copies (plain + transposed)
  __bf16* ws = (__bf16*)d_ws;
  __bf16* W1t = ws;                 // [256,2048]
  __bf16* W1b = ws + 524288;        // [2048,256]
  __bf16* W2t = ws + 1048576;
  __bf16* W2b = ws + 1114112;
  __bf16* W3t = ws + 1179648;
  __bf16* W3b = ws + 1245184;
  __bf16* W4t = ws + 1310720;
  __bf16* W4b = ws + 1376256;       // end: 1441792 elems = 2.75 MiB

  cvt_kernel<<<2048, 256, 0, stream>>>(W1, W1b, W1t, 2048, 256);
  cvt_kernel<<<256, 256, 0, stream>>>(W2, W2b, W2t, 256, 256);
  cvt_kernel<<<256, 256, 0, stream>>>(W3, W3b, W3t, 256, 256);
  cvt_kernel<<<256, 256, 0, stream>>>(W4, W4b, W4t, 256, 256);

  // alpha first (main kernel reads it to form `current`)
  biot_kernel<<<(N + 255) / 256, 256, 0, stream>>>(x, bdry, out + (long)7 * N, N, M);

  hipFuncSetAttribute(reinterpret_cast<const void*>(mlp_kernel),
                      hipFuncAttributeMaxDynamicSharedMemorySize, SMEM_BYTES);
  mlp_kernel<<<N / 32, 128, SMEM_BYTES, stream>>>(
      x, Brff, b1, b2, b3, b4, W5, b5,
      W1t, W1b, W2t, W2b, W3t, W3b, W4t, W4b, out, N);
}
